// Attention_35089882808652
// MI455X (gfx1250) — matrix operand order repaired and checked by reading, not compiled, not run on hardware
//
#include <hip/hip_runtime.h>

// ---------------------------------------------------------------------------
// CDNA5 (gfx1250) fused attention block, bf16 WMMA everywhere.
//   all matmuls: f32 inputs -> bf16 (v_cvt_pk_bf16_f32) -> v_wmma_f32_16x16x32_bf16
// Workspace layout (floats):
//   qbuf  : B*T*NH*H      = 8,388,608   (q after proj, roped+scaled in place)
//   kvtmp : B*T*2*H       = 2,097,152   (k|v proj before rope/split)
//   enc   : B*T*NH*H      = 8,388,608   (attention output)
// requires ws_size >= 75.5 MB
// ---------------------------------------------------------------------------

typedef __bf16  v16bf __attribute__((ext_vector_type(16)));
typedef __bf16  v2bf  __attribute__((ext_vector_type(2)));
typedef float   v8f   __attribute__((ext_vector_type(8)));
typedef float   v2f   __attribute__((ext_vector_type(2)));

union Frag {
    v16bf v;
    uint4 q[2];
    unsigned short u[16];
};

// two f32 -> packed bf16 (backend: v_cvt_pk_bf16_f32)
__device__ __forceinline__ v2bf cvt2(float a, float b) {
    v2f t; t.x = a; t.y = b;
    return __builtin_convertvector(t, v2bf);
}
__device__ __forceinline__ unsigned int cvt2bits(float a, float b) {
    v2bf p = cvt2(a, b);
    return __builtin_bit_cast(unsigned int, p);
}
__device__ __forceinline__ void pack4(unsigned short* d, float4 v) {
    *(v2bf*)(d + 0) = cvt2(v.x, v.y);
    *(v2bf*)(d + 2) = cvt2(v.z, v.w);
}

__device__ __forceinline__ v8f wmma_bf16(const Frag& a, const Frag& b, v8f c) {
    return __builtin_amdgcn_wmma_f32_16x16x32_bf16(
        /*neg_a=*/false, a.v, /*neg_b=*/false, b.v,
        /*c_mod=*/(short)0, c, /*reuse_a=*/false, /*reuse_b=*/false);
}

// ===========================================================================
// Generic tiled GEMM:  C[crow(m), n] = sum_k A[arow(m), k] * W(k, n)
//   logical token m in [0, M); arow(m) = (m/S)*Ta + offa + m%S
//                              crow(m) = (m/S)*Tc + offc + m%S
//   wmode 0: W row-major K x Ntot             (out-projection weights)
//   wmode 1: W head-blocked (n/256, k, n%256) (q / kv projection weights)
// Block tile 128x128x32, 256 threads (8 waves), wave tile 32x64.
// All staging addresses are hoisted out of the k-loop (pointer + stride).
// ===========================================================================
#define BM 128
#define BN 128
#define BK 32
#define LDT 40     // padded LDS stride in halves

__global__ __launch_bounds__(256)
void gemm_bf16_wmma(const float* __restrict__ A, const float* __restrict__ W,
                    float* __restrict__ C, int K, int Ntot,
                    int S, int Ta, int offa, int Tc, int offc, int wmode)
{
    __shared__ unsigned short As[BM][LDT];
    __shared__ unsigned short Bs[BN][LDT];

    const int tid  = threadIdx.x;
    const int lane = tid & 31;
    const int wid  = tid >> 5;
    const int l16  = lane & 15;
    const int half = lane >> 4;
    const int wm   = wid & 3;      // wave row  -> rows  wm*32
    const int wn   = wid >> 2;     // wave col  -> cols  wn*64
    const int bm0  = blockIdx.y * BM;
    const int bn0  = blockIdx.x * BN;

    // ---- hoisted per-thread staging pointers (k-invariant) ----
    const float*    aptr[4];
    unsigned short* adst[4];
#pragma unroll
    for (int i = 0; i < 4; ++i) {
        int l   = tid + i * 256;          // float4 slot, 0..1023
        int row = l >> 3;
        int kq  = l & 7;
        int m   = bm0 + row;
        int bb  = m / S;
        int ss  = m - bb * S;
        aptr[i] = A + (size_t)(bb * Ta + offa + ss) * K + kq * 4;
        adst[i] = &As[row][kq * 4];
    }
    const float*    wptr[4];
    unsigned short* bdst[4];
#pragma unroll
    for (int i = 0; i < 4; ++i) {
        int l  = tid + i * 256;           // float4 slot over BK x BN/4
        int kk = l >> 5;                  // 0..31
        int n  = (l & 31) * 4;            // 0..124
        int ng = bn0 + n;
        size_t addr = wmode ? ((size_t)(ng >> 8) * K + kk) * 256 + (ng & 255)
                            : (size_t)kk * Ntot + ng;
        wptr[i] = W + addr;
        bdst[i] = &Bs[n][kk];             // rows j at bdst[i] + j*LDT
    }
    const size_t wstep = wmode ? (size_t)BK * 256 : (size_t)BK * Ntot;

    const v8f vzero = {0.f,0.f,0.f,0.f,0.f,0.f,0.f,0.f};
    v8f acc[2][4];
#pragma unroll
    for (int mi = 0; mi < 2; ++mi)
#pragma unroll
        for (int ni = 0; ni < 4; ++ni) acc[mi][ni] = vzero;

    for (int k0 = 0; k0 < K; k0 += BK) {
        // ---- issue all staging loads, then convert/store ----
        float4 av[4], wv[4];
#pragma unroll
        for (int i = 0; i < 4; ++i) av[i] = *(const float4*)aptr[i];
#pragma unroll
        for (int i = 0; i < 4; ++i) wv[i] = *(const float4*)wptr[i];
#pragma unroll
        for (int i = 0; i < 4; ++i) {
            pack4(adst[i], av[i]);
            aptr[i] += BK;
            __builtin_prefetch(aptr[i], 0, 1);      // speculative, next tile
        }
#pragma unroll
        for (int i = 0; i < 4; ++i) {
            unsigned int b01 = cvt2bits(wv[i].x, wv[i].y);
            unsigned int b23 = cvt2bits(wv[i].z, wv[i].w);
            bdst[i][0 * LDT] = (unsigned short)b01;
            bdst[i][1 * LDT] = (unsigned short)(b01 >> 16);
            bdst[i][2 * LDT] = (unsigned short)b23;
            bdst[i][3 * LDT] = (unsigned short)(b23 >> 16);
            wptr[i] += wstep;
            __builtin_prefetch(wptr[i], 0, 1);
        }
        __syncthreads();

        // ---- fragments + 8 WMMAs ----
        Frag af[2], bf[4];
#pragma unroll
        for (int mi = 0; mi < 2; ++mi) {
            int row = wm * 32 + mi * 16 + l16;      // A: lane holds M=l16
            af[mi].q[0] = *(const uint4*)&As[row][half * 8];
            af[mi].q[1] = *(const uint4*)&As[row][16 + half * 8];
        }
#pragma unroll
        for (int ni = 0; ni < 4; ++ni) {
            int col = wn * 64 + ni * 16 + l16;      // B: lane holds N=l16
            bf[ni].q[0] = *(const uint4*)&Bs[col][half * 8];
            bf[ni].q[1] = *(const uint4*)&Bs[col][16 + half * 8];
        }
#pragma unroll
        for (int mi = 0; mi < 2; ++mi)
#pragma unroll
            for (int ni = 0; ni < 4; ++ni)
                acc[mi][ni] = wmma_bf16(af[mi], bf[ni], acc[mi][ni]);

        __syncthreads();
    }

    // ---- store C (layout: vgpr r -> M = r + 8*half, N = l16) ----
#pragma unroll
    for (int mi = 0; mi < 2; ++mi) {
#pragma unroll
        for (int r = 0; r < 8; ++r) {
            int row = wm * 32 + mi * 16 + r + half * 8;
            int m   = bm0 + row;
            int bb  = m / S;
            int ss  = m - bb * S;
            size_t crow = (size_t)(bb * Tc + offc + ss);
#pragma unroll
            for (int ni = 0; ni < 4; ++ni) {
                int col = bn0 + wn * 64 + ni * 16 + l16;
                C[crow * (size_t)Ntot + col] = acc[mi][ni][r];
            }
        }
    }
}

// ===========================================================================
// RoPE: q (in place, *1/16) ; k -> kout (roped) ; v -> vout (copy)
// one block per token (b*T + t), 256 threads
// ===========================================================================
__global__ __launch_bounds__(256)
void rope_kernel(float* __restrict__ qbuf, const float* __restrict__ kvtmp,
                 float* __restrict__ kout, float* __restrict__ vout,
                 const int* __restrict__ positions)
{
    const int t   = blockIdx.x;
    const int tid = threadIdx.x;
    const float pos = (float)positions[t];

    // q: 8 heads * 128 rotation pairs = 1024 pairs
#pragma unroll
    for (int i = 0; i < 4; ++i) {
        int idx = tid + i * 256;
        int h   = idx >> 7;
        int d   = idx & 127;
        float ts  = __powf(10000.0f, (float)(2 * d) * (1.0f / 256.0f));
        float rad = pos / ts;
        float s, c;
        __sincosf(rad, &s, &c);
        float* p = qbuf + (size_t)t * 2048 + h * 256;
        float a = p[d], b = p[d + 128];
        p[d]       = (a * c - b * s) * 0.0625f;      // * HEAD_DIM^-0.5
        p[d + 128] = (b * c + a * s) * 0.0625f;
    }
    // k: 128 pairs
    if (tid < 128) {
        int d = tid;
        float ts  = __powf(10000.0f, (float)(2 * d) * (1.0f / 256.0f));
        float rad = pos / ts;
        float s, c;
        __sincosf(rad, &s, &c);
        const float* kp = kvtmp + (size_t)t * 512;
        float a = kp[d], b = kp[d + 128];
        kout[(size_t)t * 256 + d]       = a * c - b * s;
        kout[(size_t)t * 256 + d + 128] = b * c + a * s;
    }
    // v copy
    vout[(size_t)t * 256 + tid] = kvtmp[(size_t)t * 512 + 256 + tid];
}

// ===========================================================================
// Flash attention (causal, MQA: 8 q-heads share one k/v)
// block: 128 threads = 4 waves; each wave owns 16 q rows; block tile 64 rows.
// key tiles of 32, staged to LDS as bf16 (K row-major, V transposed).
// Staging decomposition: l = tid + i*128  =>  key = (tid>>6)+2i, hq = (tid&63)*4,
// so one base pointer + immediate offsets covers the whole tile.
// grid: B * NH * (T/64) = 512
// ===========================================================================
#define AT_T 1024

__global__ __launch_bounds__(128)
void attn_wmma(const float* __restrict__ qbuf, const float* __restrict__ kbuf,
               const float* __restrict__ vbuf, float* __restrict__ enc)
{
    __shared__ unsigned short Ks[32][264];    // [key][h]   16.9 KB
    __shared__ unsigned short Vt[256][40];    // [h][key]   20.5 KB
    __shared__ unsigned short Ps[4][16][40];  // per-wave P  5.1 KB

    const int tid  = threadIdx.x;
    const int lane = tid & 31;
    const int wid  = tid >> 5;
    const int l16  = lane & 15;
    const int half = lane >> 4;

    const int qt    = blockIdx.x & 15;
    const int head  = (blockIdx.x >> 4) & 7;
    const int b     = blockIdx.x >> 7;
    const int qbase = qt * 64 + wid * 16;

    const v8f vzero = {0.f,0.f,0.f,0.f,0.f,0.f,0.f,0.f};

    // ---- hoisted staging pointers ----
    const int key0 = tid >> 6;                // 0 or 1
    const int hq   = (tid & 63) * 4;          // 0..252
    const float* kp = kbuf + ((size_t)(b * AT_T) + key0) * 256 + hq;
    const float* vp = vbuf + ((size_t)(b * AT_T) + key0) * 256 + hq;
    unsigned short* ksd = &Ks[key0][hq];      // slots at +i*2*264
    unsigned short* vtd = &Vt[hq][key0];      // rows j at +j*40, slots +2i

    // ---- Q fragments (A layout), f32 -> bf16, kept in registers ----
    Frag qf[8];
    {
        const float* qrow = qbuf + (size_t)(b * AT_T + qbase + l16) * 2048 + head * 256;
#pragma unroll
        for (int hc = 0; hc < 8; ++hc) {
            int lo = hc * 32 + half * 8;
            pack4(&qf[hc].u[0],  *(const float4*)(qrow + lo));
            pack4(&qf[hc].u[4],  *(const float4*)(qrow + lo + 4));
            pack4(&qf[hc].u[8],  *(const float4*)(qrow + lo + 16));
            pack4(&qf[hc].u[12], *(const float4*)(qrow + lo + 20));
        }
    }

    float mrow[8], lrow[8];
    v8f o[16];
#pragma unroll
    for (int r = 0; r < 8; ++r) { mrow[r] = -3.0e38f; lrow[r] = 0.f; }
#pragma unroll
    for (int f = 0; f < 16; ++f) o[f] = vzero;

    const int nkt = 2 * (qt + 1);            // causal: keys 0 .. qt*64+63
    for (int kt = 0; kt < nkt; ++kt) {
        const int s0 = kt * 32;
        // ---- stage K (row-major) and V (transposed) tiles, bf16 ----
#pragma unroll
        for (int i = 0; i < 16; ++i) {
            float4 kv4 = *(const float4*)(kp + i * 512);
            float4 vv4 = *(const float4*)(vp + i * 512);
            pack4(ksd + i * 2 * 264, kv4);
            unsigned int b01 = cvt2bits(vv4.x, vv4.y);
            unsigned int b23 = cvt2bits(vv4.z, vv4.w);
            vtd[i * 2 + 0 * 40] = (unsigned short)b01;
            vtd[i * 2 + 1 * 40] = (unsigned short)(b01 >> 16);
            vtd[i * 2 + 2 * 40] = (unsigned short)b23;
            vtd[i * 2 + 3 * 40] = (unsigned short)(b23 >> 16);
        }
        kp += 32 * 256;                       // next key tile
        vp += 32 * 256;
        __builtin_prefetch(kp, 0, 1);
        __builtin_prefetch(vp, 0, 1);
        __syncthreads();

        if (s0 <= qbase + 15) {              // wave has unmasked work in tile
            // ---- S = Q . K^T  (two 16x16 C tiles over 32 keys) ----
            v8f sf0 = vzero, sf1 = vzero;
#pragma unroll
            for (int hc = 0; hc < 8; ++hc) {
                int hb = hc * 32 + half * 8;
                Frag kb0, kb1;
                kb0.q[0] = *(const uint4*)&Ks[l16][hb];
                kb0.q[1] = *(const uint4*)&Ks[l16][hb + 16];
                kb1.q[0] = *(const uint4*)&Ks[16 + l16][hb];
                kb1.q[1] = *(const uint4*)&Ks[16 + l16][hb + 16];
                sf0 = wmma_bf16(qf[hc], kb0, sf0);
                sf1 = wmma_bf16(qf[hc], kb1, sf1);
            }
            // ---- causal mask + online softmax ----
#pragma unroll
            for (int r = 0; r < 8; ++r) {
                int qr = qbase + r + half * 8;
                if (s0 + l16 > qr)      sf0[r] = -3.0e38f;
                if (s0 + 16 + l16 > qr) sf1[r] = -3.0e38f;
                float tm = fmaxf(sf0[r], sf1[r]);
                tm = fmaxf(tm, __shfl_xor(tm, 1, 32));
                tm = fmaxf(tm, __shfl_xor(tm, 2, 32));
                tm = fmaxf(tm, __shfl_xor(tm, 4, 32));
                tm = fmaxf(tm, __shfl_xor(tm, 8, 32));
                float mnew = fmaxf(mrow[r], tm);
                float corr = __expf(mrow[r] - mnew);
                mrow[r] = mnew;
                float p0 = __expf(sf0[r] - mnew);
                float p1 = __expf(sf1[r] - mnew);
                sf0[r] = p0; sf1[r] = p1;
                float ps = p0 + p1;
                ps += __shfl_xor(ps, 1, 32);
                ps += __shfl_xor(ps, 2, 32);
                ps += __shfl_xor(ps, 4, 32);
                ps += __shfl_xor(ps, 8, 32);
                lrow[r] = lrow[r] * corr + ps;
#pragma unroll
                for (int f = 0; f < 16; ++f) o[f][r] *= corr;
            }
            // ---- P: C layout -> LDS -> A layout (bf16) ----
#pragma unroll
            for (int r = 0; r < 8; ++r) {
                unsigned int pb = cvt2bits(sf0[r], sf1[r]);
                Ps[wid][r + half * 8][l16]      = (unsigned short)pb;
                Ps[wid][r + half * 8][16 + l16] = (unsigned short)(pb >> 16);
            }
            asm volatile("s_wait_dscnt 0" ::: "memory");   // intra-wave LDS RAW
            Frag pa;
            pa.q[0] = *(const uint4*)&Ps[wid][l16][half * 8];
            pa.q[1] = *(const uint4*)&Ps[wid][l16][16 + half * 8];
            // ---- O += P . V  (16 head-dim column tiles) ----
#pragma unroll
            for (int f = 0; f < 16; ++f) {
                Frag vb;
                vb.q[0] = *(const uint4*)&Vt[f * 16 + l16][half * 8];
                vb.q[1] = *(const uint4*)&Vt[f * 16 + l16][16 + half * 8];
                o[f] = wmma_bf16(pa, vb, o[f]);
            }
        }
        __syncthreads();
    }

    // ---- epilogue: normalize, store encoded (B,T,NH,H) ----
#pragma unroll
    for (int r = 0; r < 8; ++r) {
        float inv = 1.0f / lrow[r];
        size_t rowoff = (size_t)(b * AT_T + qbase + r + half * 8) * 2048 + head * 256;
#pragma unroll
        for (int f = 0; f < 16; ++f)
            enc[rowoff + f * 16 + l16] = o[f][r] * inv;
    }
}

// ===========================================================================
extern "C" void kernel_launch(void* const* d_in, const int* in_sizes, int n_in,
                              void* d_out, int out_size, void* d_ws, size_t ws_size,
                              hipStream_t stream)
{
    (void)in_sizes; (void)n_in; (void)out_size; (void)ws_size;
    const float* x0        = (const float*)d_in[0];
    const float* x1        = (const float*)d_in[1];
    const float* w_q0      = (const float*)d_in[2];
    const float* w_kv0     = (const float*)d_in[3];
    const float* w_out0    = (const float*)d_in[4];
    const float* w_q1      = (const float*)d_in[5];
    const float* w_kv1     = (const float*)d_in[6];
    const float* w_out1    = (const float*)d_in[7];
    const int*   positions = (const int*)d_in[8];
    // d_in[9] = attn_mask: causal, handled analytically

    float* out = (float*)d_out;
    float* ws  = (float*)d_ws;

    float* qbuf  = ws;                 // 8,388,608 f32
    float* kvtmp = ws + 8388608;       // 2,097,152 f32
    float* enc   = ws + 10485760;      // 8,388,608 f32

    float* out0 = out;                 // (4,768,2048)
    float* out1 = out + 6291456;       // (4,256,1024)
    float* kout = out + 7340032;       // (4,1024,256)  roped k
    float* vout = out + 8388608;       // (4,1024,256)

    dim3 blk(256);
    // projections (segment rows remapped into T=1024 token space)
    gemm_bf16_wmma<<<dim3(16, 24), blk, 0, stream>>>(x0, w_q0,  qbuf,  2048, 2048, 768, 768, 0, 1024, 0,   1);
    gemm_bf16_wmma<<<dim3(4,  24), blk, 0, stream>>>(x0, w_kv0, kvtmp, 2048, 512,  768, 768, 0, 1024, 0,   1);
    gemm_bf16_wmma<<<dim3(16, 8),  blk, 0, stream>>>(x1, w_q1,  qbuf,  1024, 2048, 256, 256, 0, 1024, 768, 1);
    gemm_bf16_wmma<<<dim3(4,  8),  blk, 0, stream>>>(x1, w_kv1, kvtmp, 1024, 512,  256, 256, 0, 1024, 768, 1);
    // rope + k/v split (k,v land directly in d_out)
    rope_kernel<<<4096, 256, 0, stream>>>(qbuf, kvtmp, kout, vout, positions);
    // causal flash attention
    attn_wmma<<<512, 128, 0, stream>>>(qbuf, kout, vout, enc);
    // output projections (segment rows pulled back out of T=1024 token space)
    gemm_bf16_wmma<<<dim3(16, 24), blk, 0, stream>>>(enc, w_out0, out0, 2048, 2048, 768, 1024, 0,   768, 0, 0);
    gemm_bf16_wmma<<<dim3(8,  8),  blk, 0, stream>>>(enc, w_out1, out1, 2048, 1024, 256, 1024, 768, 256, 0, 0);
}
